// TransformerAttention_5282809774094
// MI455X (gfx1250) — matrix operand order repaired and checked by reading, not compiled, not run on hardware
//
#include <hip/hip_runtime.h>

#define BB 4
#define CC 256
#define PP 2304
#define NH 8
#define INTERC 128
#define HD 16

typedef __attribute__((ext_vector_type(16))) _Float16 v16h;
typedef __attribute__((ext_vector_type(8)))  _Float16 v8h;
typedef __attribute__((ext_vector_type(8)))  float    v8f;

// ds_swizzle butterfly within each 16-lane half (xor masks <16 never cross the
// lane15/16 boundary, matching the WMMA C/D row layout: lane = N, VGPR = M).
template<int M>
__device__ __forceinline__ float swz(float x) {
    return __int_as_float(
        __builtin_amdgcn_ds_swizzle(__float_as_int(x), (M << 10) | 0x1f));
}
__device__ __forceinline__ float rowmax16(float x) {
    x = fmaxf(x, swz<1>(x)); x = fmaxf(x, swz<2>(x));
    x = fmaxf(x, swz<4>(x)); x = fmaxf(x, swz<8>(x));
    return x;
}
__device__ __forceinline__ float rowsum16(float x) {
    x += swz<1>(x); x += swz<2>(x); x += swz<4>(x); x += swz<8>(x);
    return x;
}

// Scores live in the base-2 domain: Q is pre-scaled by (1/sqrt(16))*log2(e),
// so softmax uses bare v_exp_f32 (exp2) with no per-element multiply.
#define QSCALE (0.25f * 1.44269504088896340736f)

// ---------------------------------------------------------------------------
// Kernel 1: QKV 1x1-conv projections (f32 FMA), emit f16 in WMMA-friendly
// layouts.  Q/K: [b,h,p,16] (fragment rows contiguous).  V: [b,h,d,P]
// (contiguous along key positions).
// ---------------------------------------------------------------------------
__global__ __launch_bounds__(256)
void qkv_proj_kernel(const float* __restrict__ rgb,
                     const float* __restrict__ depth,
                     const float* __restrict__ rgbd,
                     const float* __restrict__ Wq, const float* __restrict__ bq,
                     const float* __restrict__ Wk, const float* __restrict__ bk,
                     const float* __restrict__ Wv, const float* __restrict__ bv,
                     _Float16* __restrict__ Qt, _Float16* __restrict__ Kt,
                     _Float16* __restrict__ Vh) {
    const int p = blockIdx.x * blockDim.x + threadIdx.x;   // 0..2303
    const int o = blockIdx.y;                              // 0..127
    const int z = blockIdx.z;                              // 0..11
    const int b = z & 3;
    const int t = z >> 2;                                  // 0=Q 1=K 2=V
    const float* X  = (t == 0) ? rgb : (t == 1 ? depth : rgbd);
    const float* W  = (t == 0) ? Wq  : (t == 1 ? Wk   : Wv);
    const float* Bs = (t == 0) ? bq  : (t == 1 ? bk   : bv);

    const float* wrow = W + (size_t)o * CC;
    const float* xcol = X + (size_t)b * CC * PP + p;
    float acc = Bs[o];
#pragma unroll 8
    for (int c = 0; c < CC; ++c)
        acc = fmaf(wrow[c], xcol[(size_t)c * PP], acc);

    const int h = o >> 4, d = o & 15;
    const size_t bh = (size_t)b * NH + h;
    if (t == 0) {
        Qt[(bh * PP + p) * HD + d] = (_Float16)(acc * QSCALE);
    } else if (t == 1) {
        Kt[(bh * PP + p) * HD + d] = (_Float16)acc;
    } else {
        Vh[(bh * HD + d) * PP + p] = (_Float16)acc;
    }
}

// ---------------------------------------------------------------------------
// Kernel 2: fused flash attention.  One wave32 per 16-query tile per (b,h).
// 64 keys per iteration: 4 score WMMAs (v_wmma_f32_16x16x32_f16, K padded to
// 32) + exp2-domain online softmax (ds_swizzle row reductions) + D->A probs
// transpose through a 2KB per-wave LDS tile + 2 PV WMMAs into the f32 D frag.
// ---------------------------------------------------------------------------
__global__ __launch_bounds__(32)
void flash_attn_kernel(const _Float16* __restrict__ Qt,
                       const _Float16* __restrict__ Kt,
                       const _Float16* __restrict__ Vh,
                       float* __restrict__ O) {
    __shared__ __align__(32) _Float16 pt[16 * 64];

    const int lane = threadIdx.x;      // wave32
    const int qb   = blockIdx.x * 16;  // query tile base
    const size_t bh = (size_t)blockIdx.z * NH + blockIdx.y;
    const int hl = lane >> 4;          // half-wave select
    const int ln = lane & 15;

    // A-fragment of Q^T (16x32 f16): lanes<16 hold d=0..15 (K=0..15),
    // lanes>=16 hold K=16..31 and are zero (head_dim pad).
    v16h aQ = {};
    if (hl == 0) {
        aQ = *(const v16h*)(Qt + (bh * PP + (qb + ln)) * HD);
    }

    const _Float16* krow = Kt + (bh * PP) * HD;            // + k*HD
    const _Float16* vrow = Vh + (bh * HD + ln) * PP;       // + k

    v8f acc = {};
    const v8f zc = {};
    float m[8], l[8];
#pragma unroll
    for (int j = 0; j < 8; ++j) { m[j] = -3.0e38f; l[j] = 0.0f; }

    for (int kb = 0; kb < PP; kb += 64) {
        if (kb + 64 < PP) {  // cover streaming latency of the next tile
            __builtin_prefetch(krow + (kb + 64 + ln * 4) * HD, 0, 3);
            __builtin_prefetch(vrow + kb + 64, 0, 3);
        }

        // ---- four 16x16 score tiles ----
        v16h bK0 = {}, bK1 = {}, bK2 = {}, bK3 = {};
        if (lane < 16) {
            bK0 = *(const v16h*)(krow + (kb + ln) * HD);
            bK1 = *(const v16h*)(krow + (kb + 16 + ln) * HD);
            bK2 = *(const v16h*)(krow + (kb + 32 + ln) * HD);
            bK3 = *(const v16h*)(krow + (kb + 48 + ln) * HD);
        }
        v8f s0 = __builtin_amdgcn_wmma_f32_16x16x32_f16(
            false, aQ, false, bK0, (short)0, zc, false, false);
        v8f s1 = __builtin_amdgcn_wmma_f32_16x16x32_f16(
            false, aQ, false, bK1, (short)0, zc, false, false);
        v8f s2 = __builtin_amdgcn_wmma_f32_16x16x32_f16(
            false, aQ, false, bK2, (short)0, zc, false, false);
        v8f s3 = __builtin_amdgcn_wmma_f32_16x16x32_f16(
            false, aQ, false, bK3, (short)0, zc, false, false);

        // ---- online softmax (exp2 domain) ----
        // component j = query row (j + 8*hl); lane = key column within tile.
#pragma unroll
        for (int j = 0; j < 8; ++j) {
            float rm = rowmax16(fmaxf(fmaxf(s0[j], s1[j]),
                                      fmaxf(s2[j], s3[j])));
            float mn = fmaxf(m[j], rm);
            float sc = __builtin_amdgcn_exp2f(m[j] - mn);
            float p0 = __builtin_amdgcn_exp2f(s0[j] - mn);
            float p1 = __builtin_amdgcn_exp2f(s1[j] - mn);
            float p2 = __builtin_amdgcn_exp2f(s2[j] - mn);
            float p3 = __builtin_amdgcn_exp2f(s3[j] - mn);
            l[j] = l[j] * sc + rowsum16((p0 + p1) + (p2 + p3));
            acc[j] *= sc;
            m[j] = mn;
            // stage probs [q_local][k_local 0..63] for the D->A transpose
            _Float16* row = pt + (j + hl * 8) * 64 + ln;
            row[0]  = (_Float16)p0;
            row[16] = (_Float16)p1;
            row[32] = (_Float16)p2;
            row[48] = (_Float16)p3;
        }

        // ---- two PV WMMAs over the 16x64 prob tile ----
        // (same-wave LDS ops are pipeline-ordered; no barrier needed)
#pragma unroll
        for (int kk = 0; kk < 64; kk += 32) {
            v16h aP;
            {
                const int c = hl * 16;
                const v8h a0 = *(const v8h*)(pt + ln * 64 + kk + c);
                const v8h a1 = *(const v8h*)(pt + ln * 64 + kk + 8 + c);
#pragma unroll
                for (int i = 0; i < 8; ++i) { aP[i] = a0[i]; aP[i + 8] = a1[i]; }
            }
            // V B-fragment (32 keys x 16 d): lane = d, halves = 16 contig k.
            const v16h bV = *(const v16h*)(vrow + kb + kk + hl * 16);
            acc = __builtin_amdgcn_wmma_f32_16x16x32_f16(
                false, aP, false, bV, (short)0, acc, false, false);
        }
    }

    // Normalize and store O[b,h,q,d] (fp32).
#pragma unroll
    for (int j = 0; j < 8; ++j) {
        const float inv = 1.0f / l[j];
        O[(bh * PP + (qb + j + hl * 8)) * HD + ln] = acc[j] * inv;
    }
}

// ---------------------------------------------------------------------------
// Kernel 3: output projection (Wo x O + bo) fused with the three residual
// adds.  O is 4.7 MB -> L2 resident, so the d-strided reads are cheap.
// ---------------------------------------------------------------------------
__global__ __launch_bounds__(256)
void out_proj_kernel(const float* __restrict__ rgb,
                     const float* __restrict__ depth,
                     const float* __restrict__ rgbd,
                     const float* __restrict__ Wo, const float* __restrict__ bo,
                     const float* __restrict__ O, float* __restrict__ out) {
    const int p  = blockIdx.x * blockDim.x + threadIdx.x;
    const int co = blockIdx.y;
    const int b  = blockIdx.z;
    const float* wrow = Wo + (size_t)co * INTERC;
    float r = bo[co];
#pragma unroll
    for (int h = 0; h < NH; ++h) {
        const float4* ov =
            (const float4*)(O + (((size_t)b * NH + h) * PP + p) * HD);
        const float* wh = wrow + h * HD;
#pragma unroll
        for (int d4 = 0; d4 < 4; ++d4) {
            float4 o4 = ov[d4];
            r = fmaf(wh[d4 * 4 + 0], o4.x, r);
            r = fmaf(wh[d4 * 4 + 1], o4.y, r);
            r = fmaf(wh[d4 * 4 + 2], o4.z, r);
            r = fmaf(wh[d4 * 4 + 3], o4.w, r);
        }
    }
    const size_t idx = ((size_t)b * CC + co) * PP + p;
    const size_t N = (size_t)BB * CC * PP;
    out[idx]         = rgb[idx]   + r;
    out[N + idx]     = depth[idx] + r;
    out[2 * N + idx] = rgbd[idx]  + r;
}

extern "C" void kernel_launch(void* const* d_in, const int* in_sizes, int n_in,
                              void* d_out, int out_size, void* d_ws,
                              size_t ws_size, hipStream_t stream) {
    (void)in_sizes; (void)n_in; (void)out_size; (void)ws_size;
    const float* rgb   = (const float*)d_in[0];
    const float* depth = (const float*)d_in[1];
    const float* rgbd  = (const float*)d_in[2];
    const float* Wq = (const float*)d_in[3];
    const float* bq = (const float*)d_in[4];
    const float* Wk = (const float*)d_in[5];
    const float* bk = (const float*)d_in[6];
    const float* Wv = (const float*)d_in[7];
    const float* bv = (const float*)d_in[8];
    const float* Wo = (const float*)d_in[9];
    const float* bo = (const float*)d_in[10];
    float* out = (float*)d_out;

    // Workspace: Qt | Kt | Vh (f16) | O (f32)  -> ~11.8 MB total
    const size_t nQKV = (size_t)BB * NH * PP * HD;  // 1,179,648 elements
    _Float16* Qt = (_Float16*)d_ws;
    _Float16* Kt = Qt + nQKV;
    _Float16* Vh = Kt + nQKV;
    float*    O  = (float*)(Vh + nQKV);

    dim3 pb(256), pg(PP / 256, INTERC, 12);
    qkv_proj_kernel<<<pg, pb, 0, stream>>>(rgb, depth, rgbd, Wq, bq, Wk, bk,
                                           Wv, bv, Qt, Kt, Vh);

    dim3 ab(32), ag(PP / 16, NH, BB);
    flash_attn_kernel<<<ag, ab, 0, stream>>>(Qt, Kt, Vh, O);

    dim3 ob(256), og(PP / 256, CC, BB);
    out_proj_kernel<<<og, ob, 0, stream>>>(rgb, depth, rgbd, Wo, bo, O, out);
}
